// ProteinMPNNNodeUpdate_40450001994150
// MI455X (gfx1250) — compile-verified, matrix operand order checked by reading, not run on hardware
//
#include <hip/hip_runtime.h>

typedef __attribute__((ext_vector_type(16))) _Float16 v16h;
typedef __attribute__((ext_vector_type(8)))  float    v8f;

#define NN 2048
#define KK 48
#define CC 128

// ---------- helpers ----------
// tanh-form GELU: ~8 VALU ops using hardware exp/rcp, overflow-safe.
__device__ __forceinline__ float gelu_fast(float x) {
  float t  = x * (0.7978845608028654f + 0.0356774081363f * x * x);
  float e  = __expf(2.0f * t);
  float th = 1.0f - 2.0f / (1.0f + e);
  return 0.5f * x * (1.0f + th);
}

// Build a 16x32 f16 A-fragment from an fp32 row segment.
// p points at rowbase + kbase (kbase = 0 for lanes 0-15, 8 for lanes 16-31).
// Element e (0..7) -> K = kbase+e ; e (8..15) -> K = kbase+16+(e-8).
__device__ __forceinline__ v16h load_a_f32(const float* p) {
  const float4* q0 = (const float4*)p;
  float4 x0 = q0[0], x1 = q0[1];
  const float4* q1 = (const float4*)(p + 16);
  float4 x2 = q1[0], x3 = q1[1];
  v16h r;
  r[0] = (_Float16)x0.x;  r[1] = (_Float16)x0.y;
  r[2] = (_Float16)x0.z;  r[3] = (_Float16)x0.w;
  r[4] = (_Float16)x1.x;  r[5] = (_Float16)x1.y;
  r[6] = (_Float16)x1.z;  r[7] = (_Float16)x1.w;
  r[8] = (_Float16)x2.x;  r[9] = (_Float16)x2.y;
  r[10] = (_Float16)x2.z; r[11] = (_Float16)x2.w;
  r[12] = (_Float16)x3.x; r[13] = (_Float16)x3.y;
  r[14] = (_Float16)x3.z; r[15] = (_Float16)x3.w;
  return r;
}

#define WMMA(A, B, C) __builtin_amdgcn_wmma_f32_16x16x32_f16( \
    false, (A), false, (B), (short)0, (C), false, false)

// ---------- pre-pass: convert W1/W2/W3 fp32 -> f16 B-fragment layout ----------
// frag index: (((ks*8 + nt)*32 + lane)*16 + e)
// source element: k = ks*32 + kbase + (e<8 ? e : e+8), n = nt*16 + (lane&15)
__global__ __launch_bounds__(256) void swizzle_weights(
    const float* __restrict__ W1, const float* __restrict__ W2,
    const float* __restrict__ W3, _Float16* __restrict__ ws) {
  int i = blockIdx.x * 256 + threadIdx.x;
  const float* src;
  int idx;
  if (i < 49152)        { idx = i;          src = W1; }
  else if (i < 65536)   { idx = i - 49152;  src = W2; }
  else if (i < 81920)   { idx = i - 65536;  src = W3; }
  else return;
  int e    = idx & 15;
  int lane = (idx >> 4) & 31;
  int nt   = (idx >> 9) & 7;
  int ks   = idx >> 12;
  int kbase = (lane & 16) ? 8 : 0;
  int eoff  = (e < 8) ? e : (e + 8);
  int k = ks * 32 + kbase + eoff;
  int n = nt * 16 + (lane & 15);
  ws[i] = (_Float16)src[k * CC + n];
}

// ---------- main kernel: one wave per node, all 3 edge-row tiles in flight ----
__global__ __launch_bounds__(256) void mpnn_node_update(
    const float* __restrict__ node_h, const float* __restrict__ edge_h,
    const int* __restrict__ edge_index,
    const float* __restrict__ b1, const float* __restrict__ b2,
    const float* __restrict__ b3, const float* __restrict__ Wu,
    const float* __restrict__ bu, const float* __restrict__ ln_g,
    const float* __restrict__ ln_b, const _Float16* __restrict__ wfrag,
    float* __restrict__ out) {
  __shared__ __align__(32) _Float16 sW1[49152];   // 96 KB
  __shared__ __align__(32) _Float16 sW2[16384];   // 32 KB
  __shared__ float sB1[128];
  __shared__ float sB2[128];
  __shared__ __align__(32) _Float16 sA2[8][6144]; // 3 tiles per wave, 96 KB
  __shared__ float sY[8][128];                    // 4 KB
  // total ~229 KB of the 320 KB WGP LDS

  const int tid = threadIdx.x;

  // cooperative copy of pre-swizzled f16 W1/W2 fragments into LDS
  {
    const uint4* s  = (const uint4*)wfrag;
    uint4* d1 = (uint4*)sW1;
    uint4* d2 = (uint4*)sW2;
    #pragma unroll 1
    for (int i = tid; i < 6144; i += 256) d1[i] = s[i];
    #pragma unroll 1
    for (int i = tid; i < 2048; i += 256) d2[i] = s[6144 + i];
  }
  if (tid < 128) sB1[tid] = b1[tid];
  else           sB2[tid - 128] = b2[tid - 128];
  __syncthreads();

  const int wave   = tid >> 5;
  const int lane   = tid & 31;
  const int lane15 = lane & 15;
  const int kbase  = (lane & 16) ? 8 : 0;
  const int g  = blockIdx.x * 8 + wave;  // node id in [0, B*N)
  const int bb = g >> 11;                // batch index (N = 2048)

  const float* irow  = node_h + (size_t)g * CC;
  const float* irowk = irow + kbase;
  // one base pointer for all 3 edge tiles: tile t at +t*2048 floats (imm offset)
  const float* ebasek = edge_h + ((size_t)g * KK + lane15) * CC + kbase;
  _Float16* a2 = &sA2[wave][0];
  const int hi   = (lane >> 3) & 1;
  const int mofs = (lane & 16) ? 8 : 0;

  // gathered neighbor row pointers (kbase folded in)
  const float* jrowk[3];
  #pragma unroll
  for (int t = 0; t < 3; ++t) {
    int nbr = edge_index[(size_t)g * KK + t * 16 + lane15];
    jrowk[t] = node_h + ((size_t)(bb << 11) + nbr) * CC + kbase;
  }

  const _Float16* w1p  = sW1 + lane * 16;           // frags 0..63 (offsets < 64 KB)
  const _Float16* w1hi = w1p + 64 * 512;            // frags 64..95
  const _Float16* w2p  = sW2 + lane * 16;
  const _Float16* w3g  = wfrag + 65536 + lane * 16; // W3 frags straight from L2

  // compile-time fragment loads (split base keeps DS imm offsets < 64 KB)
  auto LDW1 = [&](int i) -> v16h {
    return (i < 64) ? *(const v16h*)(w1p + i * 512)
                    : *(const v16h*)(w1hi + (i - 64) * 512);
  };
  auto LDW2 = [&](int i) -> v16h { return *(const v16h*)(w2p + i * 512); };
  auto LDW3 = [&](int i) -> v16h { return *(const v16h*)(w3g + i * 512); };

  v8f acc0[8], acc1[8], acc2[8];

  // ---- GEMM1 phase A: node_i columns (ks 0..3) identical for all 48 edges ----
  #pragma unroll
  for (int nt = 0; nt < 8; ++nt) {
    float bv = sB1[nt * 16 + lane15];
    #pragma unroll
    for (int r = 0; r < 8; ++r) acc0[nt][r] = bv;
  }
  {
    v16h bq[4];
    bq[0] = LDW1(0);
    bq[1] = LDW1(1);
    v16h a_cur = load_a_f32(irowk);
    #pragma unroll
    for (int ks = 0; ks < 4; ++ks) {
      v16h a_nxt;
      if (ks < 3) a_nxt = load_a_f32(irowk + (ks + 1) * 32);
      #pragma unroll
      for (int nt = 0; nt < 8; ++nt) {
        const int cur = ks * 8 + nt;
        if (cur + 2 < 32) bq[(cur + 2) & 3] = LDW1(cur + 2);
        acc0[nt] = WMMA(a_cur, bq[cur & 3], acc0[nt]);
      }
      if (ks < 3) a_cur = a_nxt;
    }
  }
  #pragma unroll
  for (int nt = 0; nt < 8; ++nt) { acc1[nt] = acc0[nt]; acc2[nt] = acc0[nt]; }

  // ---- GEMM1 phase B: node_j (ks 4..7) and edge_h (ks 8..11) columns ----
  // each B fragment feeds 3 WMMAs -> 3x less LDS weight traffic
  {
    v16h bq[4];
    bq[32 & 3] = LDW1(32);
    bq[33 & 3] = LDW1(33);
    #pragma unroll
    for (int ks = 4; ks < 12; ++ks) {
      v16h af0, af1, af2;
      if (ks < 8) {
        af0 = load_a_f32(jrowk[0] + (ks - 4) * 32);
        af1 = load_a_f32(jrowk[1] + (ks - 4) * 32);
        af2 = load_a_f32(jrowk[2] + (ks - 4) * 32);
      } else {
        af0 = load_a_f32(ebasek + (ks - 8) * 32);
        af1 = load_a_f32(ebasek + 2048 + (ks - 8) * 32);
        af2 = load_a_f32(ebasek + 4096 + (ks - 8) * 32);
      }
      #pragma unroll
      for (int nt = 0; nt < 8; ++nt) {
        const int cur = ks * 8 + nt;
        if (cur + 2 < 96) bq[(cur + 2) & 3] = LDW1(cur + 2);
        acc0[nt] = WMMA(af0, bq[cur & 3], acc0[nt]);
        acc1[nt] = WMMA(af1, bq[cur & 3], acc1[nt]);
        acc2[nt] = WMMA(af2, bq[cur & 3], acc2[nt]);
      }
    }
  }
  // GELU -> LDS staging in A-fragment layout (3 tiles)
  #pragma unroll
  for (int nt = 0; nt < 8; ++nt) {
    int base = (nt >> 1) * 512 + hi * 256 + (lane & 7) + (nt & 1) * 8;
    #pragma unroll
    for (int r = 0; r < 8; ++r) {
      int off = base + (r + mofs) * 16;
      a2[off]        = (_Float16)gelu_fast(acc0[nt][r]);
      a2[off + 2048] = (_Float16)gelu_fast(acc1[nt][r]);
      a2[off + 4096] = (_Float16)gelu_fast(acc2[nt][r]);
    }
  }
  asm volatile("s_wait_dscnt 0" ::: "memory");

  // ---- GEMM2: (16 x 128) x (128 x 128), 3 tiles share each B fragment ----
  #pragma unroll
  for (int nt = 0; nt < 8; ++nt) {
    float bv = sB2[nt * 16 + lane15];
    #pragma unroll
    for (int r = 0; r < 8; ++r) acc0[nt][r] = bv;
    acc1[nt] = acc0[nt];
    acc2[nt] = acc0[nt];
  }
  {
    v16h bq[4];
    bq[0] = LDW2(0);
    bq[1] = LDW2(1);
    #pragma unroll
    for (int ks = 0; ks < 4; ++ks) {
      v16h af0 = *(const v16h*)&a2[ks * 512 + lane * 16];
      v16h af1 = *(const v16h*)&a2[2048 + ks * 512 + lane * 16];
      v16h af2 = *(const v16h*)&a2[4096 + ks * 512 + lane * 16];
      #pragma unroll
      for (int nt = 0; nt < 8; ++nt) {
        const int cur = ks * 8 + nt;
        if (cur + 2 < 32) bq[(cur + 2) & 3] = LDW2(cur + 2);
        acc0[nt] = WMMA(af0, bq[cur & 3], acc0[nt]);
        acc1[nt] = WMMA(af1, bq[cur & 3], acc1[nt]);
        acc2[nt] = WMMA(af2, bq[cur & 3], acc2[nt]);
      }
    }
  }
  #pragma unroll
  for (int nt = 0; nt < 8; ++nt) {
    int base = (nt >> 1) * 512 + hi * 256 + (lane & 7) + (nt & 1) * 8;
    #pragma unroll
    for (int r = 0; r < 8; ++r) {
      int off = base + (r + mofs) * 16;
      a2[off]        = (_Float16)gelu_fast(acc0[nt][r]);
      a2[off + 2048] = (_Float16)gelu_fast(acc1[nt][r]);
      a2[off + 4096] = (_Float16)gelu_fast(acc2[nt][r]);
    }
  }
  asm volatile("s_wait_dscnt 0" ::: "memory");

  // ---- GEMM3: B fragments streamed from global (L2-resident, broadcast) ----
  #pragma unroll
  for (int nt = 0; nt < 8; ++nt)
    #pragma unroll
    for (int r = 0; r < 8; ++r) {
      acc0[nt][r] = 0.0f; acc1[nt][r] = 0.0f; acc2[nt][r] = 0.0f;
    }
  {
    v16h bq[4];
    bq[0] = LDW3(0);
    bq[1] = LDW3(1);
    #pragma unroll
    for (int ks = 0; ks < 4; ++ks) {
      v16h af0 = *(const v16h*)&a2[ks * 512 + lane * 16];
      v16h af1 = *(const v16h*)&a2[2048 + ks * 512 + lane * 16];
      v16h af2 = *(const v16h*)&a2[4096 + ks * 512 + lane * 16];
      #pragma unroll
      for (int nt = 0; nt < 8; ++nt) {
        const int cur = ks * 8 + nt;
        if (cur + 2 < 32) bq[(cur + 2) & 3] = LDW3(cur + 2);
        acc0[nt] = WMMA(af0, bq[cur & 3], acc0[nt]);
        acc1[nt] = WMMA(af1, bq[cur & 3], acc1[nt]);
        acc2[nt] = WMMA(af2, bq[cur & 3], acc2[nt]);
      }
    }
  }

  // ---- reduce messages over rows and tiles ----
  float msgs[8];
  #pragma unroll
  for (int nt = 0; nt < 8; ++nt) {
    float s = 0.f;
    #pragma unroll
    for (int r = 0; r < 8; ++r) s += acc0[nt][r] + acc1[nt][r] + acc2[nt][r];
    msgs[nt] = s;
  }

  // ---- combine halves, add 48*b3, /30, LayerNorm #1 ----
  float u[8];
  float sum = 0.f, sq = 0.f;
  #pragma unroll
  for (int nt = 0; nt < 8; ++nt) {
    float s = msgs[nt];
    s += __shfl_xor(s, 16, 32);  // combine row halves
    int col = nt * 16 + lane15;
    float val = irow[col] + (s + 48.0f * b3[col]) * (1.0f / 30.0f);
    u[nt] = val;
    sum += val;
    sq  += val * val;
  }
  sum += __shfl_xor(sum, 8, 32);  sq += __shfl_xor(sq, 8, 32);
  sum += __shfl_xor(sum, 4, 32);  sq += __shfl_xor(sq, 4, 32);
  sum += __shfl_xor(sum, 2, 32);  sq += __shfl_xor(sq, 2, 32);
  sum += __shfl_xor(sum, 1, 32);  sq += __shfl_xor(sq, 1, 32);
  float mean = sum * (1.0f / 128.0f);
  float var  = sq * (1.0f / 128.0f) - mean * mean;
  float rstd = rsqrtf(var + 1e-5f);

  if (lane < 16) {
    #pragma unroll
    for (int nt = 0; nt < 8; ++nt) {
      int col = nt * 16 + lane;
      sY[wave][col] = (u[nt] - mean) * rstd * ln_g[col] + ln_b[col];
    }
  }
  asm volatile("s_wait_dscnt 0" ::: "memory");

  // ---- node update GEMM (fp32 VALU): y @ Wu + bu ; 4 columns per lane ----
  const int c0 = lane * 4;
  float4 accu = *(const float4*)(bu + c0);
  #pragma unroll 8
  for (int j = 0; j < 128; ++j) {
    float yv = sY[wave][j];
    float4 w = *(const float4*)(Wu + (size_t)j * CC + c0);
    accu.x += yv * w.x;
    accu.y += yv * w.y;
    accu.z += yv * w.z;
    accu.w += yv * w.w;
  }
  float4 nh = *(const float4*)(irow + c0);
  float v0 = nh.x + accu.x, v1 = nh.y + accu.y;
  float v2 = nh.z + accu.z, v3 = nh.w + accu.w;
  float s2 = v0 + v1 + v2 + v3;
  float q2 = v0 * v0 + v1 * v1 + v2 * v2 + v3 * v3;
  s2 += __shfl_xor(s2, 16, 32);  q2 += __shfl_xor(q2, 16, 32);
  s2 += __shfl_xor(s2, 8, 32);   q2 += __shfl_xor(q2, 8, 32);
  s2 += __shfl_xor(s2, 4, 32);   q2 += __shfl_xor(q2, 4, 32);
  s2 += __shfl_xor(s2, 2, 32);   q2 += __shfl_xor(q2, 2, 32);
  s2 += __shfl_xor(s2, 1, 32);   q2 += __shfl_xor(q2, 1, 32);
  float mean2 = s2 * (1.0f / 128.0f);
  float var2  = q2 * (1.0f / 128.0f) - mean2 * mean2;
  float rstd2 = rsqrtf(var2 + 1e-5f);
  float4 gg = *(const float4*)(ln_g + c0);
  float4 bl = *(const float4*)(ln_b + c0);
  float4 o;
  o.x = (v0 - mean2) * rstd2 * gg.x + bl.x;
  o.y = (v1 - mean2) * rstd2 * gg.y + bl.y;
  o.z = (v2 - mean2) * rstd2 * gg.z + bl.z;
  o.w = (v3 - mean2) * rstd2 * gg.w + bl.w;
  *(float4*)(out + (size_t)g * CC + c0) = o;
}

extern "C" void kernel_launch(void* const* d_in, const int* in_sizes, int n_in,
                              void* d_out, int out_size, void* d_ws, size_t ws_size,
                              hipStream_t stream) {
  const float* node_h     = (const float*)d_in[0];
  const float* edge_h     = (const float*)d_in[1];
  const int*   edge_index = (const int*)d_in[2];
  const float* W1 = (const float*)d_in[3];
  const float* b1 = (const float*)d_in[4];
  const float* W2 = (const float*)d_in[5];
  const float* b2 = (const float*)d_in[6];
  const float* W3 = (const float*)d_in[7];
  const float* b3 = (const float*)d_in[8];
  const float* Wu = (const float*)d_in[9];
  const float* bu = (const float*)d_in[10];
  const float* ln_g = (const float*)d_in[11];
  const float* ln_b = (const float*)d_in[12];
  _Float16* wfrag = (_Float16*)d_ws;  // 81920 halfs = 160 KB of scratch

  swizzle_weights<<<320, 256, 0, stream>>>(W1, W2, W3, wfrag);
  // B*N = 4096 nodes, 8 nodes (waves) per 256-thread block
  mpnn_node_update<<<512, 256, 0, stream>>>(node_h, edge_h, edge_index,
                                            b1, b2, b3, Wu, bu, ln_g, ln_b,
                                            wfrag, (float*)d_out);
}
